// Compress_26654567039062
// MI455X (gfx1250) — matrix operand order repaired and checked by reading, not compiled, not run on hardware
//
#include <hip/hip_runtime.h>
#include <hip/hip_bf16.h>

typedef __bf16 bf16_t;
typedef __attribute__((ext_vector_type(16))) __bf16 v16bf;
typedef __attribute__((ext_vector_type(8)))  float  v8f;

#define H_DIM   2048
#define NH_     32
#define NKV_    8
#define HD_     64
#define T_LEN   2048
#define S_LEN   1024
#define TS_LEN  3072      // T + S
#define I_DIM   5632

#define BPAD    40        // LDS B-tile row stride in bf16 (80B: bank-conflict-free)

union FragU { float4 f4[2]; v16bf v; };

// ---- CDNA5 async global->LDS (ASYNCcnt) helpers -----------------------------
__device__ __forceinline__ uint32_t lds_off(const void* p) {
  // flat LDS addresses carry the LDS byte offset in the low 32 bits
  return (uint32_t)(uintptr_t)p;
}
__device__ __forceinline__ void async_load_b128(const bf16_t* gptr, uint32_t loff) {
  asm volatile("global_load_async_to_lds_b128 %0, %1, off"
               :: "v"(loff), "v"(gptr) : "memory");
}
__device__ __forceinline__ void wait_async_le1() {
  asm volatile("s_wait_asynccnt 0x1" ::: "memory");
}
__device__ __forceinline__ void wait_async_le0() {
  asm volatile("s_wait_asynccnt 0x0" ::: "memory");
}

// A-fragment (16x32 bf16, M x K): lane = (row = lane&15, half = lane>>4)
// element e -> K = 16*half + e
__device__ __forceinline__ v16bf load_a_frag(const bf16_t* A, int lda, int m0, int k0, int lane) {
  int row = lane & 15, half = lane >> 4;
  const bf16_t* p = A + (size_t)(m0 + row) * lda + k0 + half * 16;
  FragU u;
  u.f4[0] = *reinterpret_cast<const float4*>(p);
  u.f4[1] = *reinterpret_cast<const float4*>(p + 8);
  return u.v;
}

// B-fragment (32x16 bf16, K x N) from row-major W[N][K]:
// lane = (col = lane&15, half = lane>>4); element e -> K = 16*half + e
__device__ __forceinline__ v16bf load_b_frag(const bf16_t* W, int ldk, int n0, int k0, int lane) {
  int col = lane & 15, half = lane >> 4;
  const bf16_t* p = W + (size_t)(n0 + col) * ldk + k0 + half * 16;
  FragU u;
  u.f4[0] = *reinterpret_cast<const float4*>(p);
  u.f4[1] = *reinterpret_cast<const float4*>(p + 8);
  return u.v;
}

// B-fragment from an LDS tile [64][BPAD] holding a 64(N) x 32(K) chunk
__device__ __forceinline__ v16bf load_b_frag_lds(const bf16_t* Bl, int n0, int lane) {
  int col = lane & 15, half = lane >> 4;
  const bf16_t* p = Bl + (n0 + col) * BPAD + half * 16;
  FragU u;
  u.f4[0] = *reinterpret_cast<const float4*>(p);
  u.f4[1] = *reinterpret_cast<const float4*>(p + 8);
  return u.v;
}

// ---------------------------------------------------------------------------
// NT GEMM: C[M,N] = A[M,K] * W[N,K]^T  (+ optional fp32 residual add)
// block = 256 threads = 8 waves; block tile = 128(M) x 64(N); wave = 16 x 64
// B tile (64x32) staged in LDS via async DMA, double-buffered.
// Requires M % 128 == 0, N % 64 == 0, K % 32 == 0.
// ---------------------------------------------------------------------------
template<bool ADD, bool OUT_BF16>
__global__ void __launch_bounds__(256)
gemm_nt(const bf16_t* __restrict__ A, const bf16_t* __restrict__ W,
        const float* __restrict__ addsrc, void* __restrict__ Cout,
        int M, int N, int K) {
  __shared__ __align__(16) bf16_t Bl[2][64 * BPAD];

  int lane = threadIdx.x & 31;
  int wave = threadIdx.x >> 5;
  int m0 = blockIdx.y * 128 + wave * 16;
  int n0 = blockIdx.x * 64;

  // staging map: thread t -> B row (t>>2), 16B segment (t&3)
  int brow = threadIdx.x >> 2;
  int bseg = (threadIdx.x & 3) * 8;
  const bf16_t* wsrc = W + (size_t)(n0 + brow) * K + bseg;
  uint32_t ldst0 = lds_off(&Bl[0][brow * BPAD + bseg]);
  uint32_t ldst1 = lds_off(&Bl[1][brow * BPAD + bseg]);

  async_load_b128(wsrc, ldst0);

  v8f acc[4] = {};
  int nk = K >> 5;
  for (int ki = 0; ki < nk; ++ki) {
    int cur = ki & 1;
    if (ki + 1 < nk) {
      async_load_b128(wsrc + (size_t)(ki + 1) * 32, cur ? ldst0 : ldst1);
      wait_async_le1();              // current buffer's DMA complete
    } else {
      wait_async_le0();
    }
    __syncthreads();

    __builtin_prefetch(A + (size_t)m0 * K + ki * 32 + 256, 0, 1);  // global_prefetch_b8
    v16bf a = load_a_frag(A, K, m0, ki * 32, lane);
#pragma unroll
    for (int j = 0; j < 4; ++j) {
      v16bf b = load_b_frag_lds(&Bl[cur][0], j * 16, lane);
      acc[j] = __builtin_amdgcn_wmma_f32_16x16x32_bf16(
          false, a, false, b, (short)0, acc[j], false, false);
    }
    __syncthreads();                 // all waves done with Bl[cur] before overwrite
  }

  int col = lane & 15, half = lane >> 4;
#pragma unroll
  for (int j = 0; j < 4; ++j) {
#pragma unroll
    for (int r = 0; r < 8; ++r) {
      int m = m0 + r + 8 * half;
      int n = n0 + j * 16 + col;
      float v = acc[j][r];
      if (ADD) v += addsrc[(size_t)m * N + n];
      if (OUT_BF16) ((bf16_t*)Cout)[(size_t)m * N + n] = (bf16_t)v;
      else          ((float*)Cout)[(size_t)m * N + n]  = v;
    }
  }
}

// ---------------------------------------------------------------------------
// Flash attention, GQA 4:1.  One wave per (16-query tile, head).
// q: [S, H]  k: [TS, 512]  vT: [512, TS]  o: [S, H]   all bf16
// ---------------------------------------------------------------------------
__global__ void __launch_bounds__(32)
attn_kernel(const bf16_t* __restrict__ q, const bf16_t* __restrict__ kbuf,
            const bf16_t* __restrict__ vT, bf16_t* __restrict__ o) {
  __shared__ __align__(16) bf16_t p_lds[16 * 32];
  int lane = threadIdx.x & 31;
  int s0 = blockIdx.x * 16;
  int h  = blockIdx.y;
  int kcol0 = (h >> 2) * HD_;                 // kv head base column
  const float scale = 0.125f;                 // HD^-0.5

  v16bf aq0 = load_a_frag(q, H_DIM, s0, h * HD_ + 0,  lane);
  v16bf aq1 = load_a_frag(q, H_DIM, s0, h * HD_ + 32, lane);

  float m_run[8], l_run[8];
#pragma unroll
  for (int r = 0; r < 8; ++r) { m_run[r] = -3.0e38f; l_run[r] = 0.f; }
  v8f oacc[4] = {};

  int col = lane & 15, half = lane >> 4;

  for (int t0 = 0; t0 < TS_LEN; t0 += 32) {
    v8f c_lo = {}, c_hi = {};
    v16bf bk;
    bk = load_b_frag(kbuf, NKV_ * HD_, t0,      kcol0 + 0,  lane);
    c_lo = __builtin_amdgcn_wmma_f32_16x16x32_bf16(false, aq0, false, bk, (short)0, c_lo, false, false);
    bk = load_b_frag(kbuf, NKV_ * HD_, t0,      kcol0 + 32, lane);
    c_lo = __builtin_amdgcn_wmma_f32_16x16x32_bf16(false, aq1, false, bk, (short)0, c_lo, false, false);
    bk = load_b_frag(kbuf, NKV_ * HD_, t0 + 16, kcol0 + 0,  lane);
    c_hi = __builtin_amdgcn_wmma_f32_16x16x32_bf16(false, aq0, false, bk, (short)0, c_hi, false, false);
    bk = load_b_frag(kbuf, NKV_ * HD_, t0 + 16, kcol0 + 32, lane);
    c_hi = __builtin_amdgcn_wmma_f32_16x16x32_bf16(false, aq1, false, bk, (short)0, c_hi, false, false);

    // online softmax per row (row r+8*half lives in vgpr r; 16 lanes = 16 keys)
#pragma unroll
    for (int r = 0; r < 8; ++r) {
      float x0 = c_lo[r] * scale;
      float x1 = c_hi[r] * scale;
      float mx = fmaxf(x0, x1);
#pragma unroll
      for (int off = 1; off < 16; off <<= 1) mx = fmaxf(mx, __shfl_xor(mx, off, 32));
      float mnew = fmaxf(m_run[r], mx);
      float p0 = __expf(x0 - mnew);
      float p1 = __expf(x1 - mnew);
      float rs = p0 + p1;
#pragma unroll
      for (int off = 1; off < 16; off <<= 1) rs += __shfl_xor(rs, off, 32);
      float alpha = __expf(m_run[r] - mnew);
      m_run[r] = mnew;
      l_run[r] = l_run[r] * alpha + rs;
#pragma unroll
      for (int j = 0; j < 4; ++j) oacc[j][r] *= alpha;
      int m = r + 8 * half;
      p_lds[m * 32 + col]      = (bf16_t)p0;   // key = col      (c_lo)
      p_lds[m * 32 + 16 + col] = (bf16_t)p1;   // key = col + 16 (c_hi)
    }

    // P (16x32) as A-fragment via LDS transpose; V chunk as B-fragments
    {
      int row = lane & 15;
      const bf16_t* p = &p_lds[row * 32 + half * 16];
      FragU u;
      u.f4[0] = *reinterpret_cast<const float4*>(p);
      u.f4[1] = *reinterpret_cast<const float4*>(p + 8);
      v16bf pf = u.v;
#pragma unroll
      for (int j = 0; j < 4; ++j) {
        v16bf bv = load_b_frag(vT + (size_t)kcol0 * TS_LEN, TS_LEN, j * 16, t0, lane);
        oacc[j] = __builtin_amdgcn_wmma_f32_16x16x32_bf16(
            false, pf, false, bv, (short)0, oacc[j], false, false);
      }
    }
  }

#pragma unroll
  for (int j = 0; j < 4; ++j) {
#pragma unroll
    for (int r = 0; r < 8; ++r) {
      int m = s0 + r + 8 * half;
      int d = j * 16 + col;
      o[(size_t)m * H_DIM + h * HD_ + d] = (bf16_t)(oacc[j][r] / l_run[r]);
    }
  }
}

// ---------------------------------------------------------------------------
// RMS norm: one 256-thread block per row; bf16 out (+ optional fp32 copy)
// ---------------------------------------------------------------------------
template<bool F32OUT>
__global__ void __launch_bounds__(256)
rmsnorm_kernel(const float* __restrict__ in, const float* __restrict__ w,
               bf16_t* __restrict__ out_bf, float* __restrict__ out_f) {
  __shared__ float red[8];
  int row = blockIdx.x;
  const float* x = in + (size_t)row * H_DIM;
  float ss = 0.f;
  for (int i = threadIdx.x; i < H_DIM; i += 256) { float v = x[i]; ss += v * v; }
#pragma unroll
  for (int off = 1; off < 32; off <<= 1) ss += __shfl_xor(ss, off, 32);
  if ((threadIdx.x & 31) == 0) red[threadIdx.x >> 5] = ss;
  __syncthreads();
  float tot = 0.f;
#pragma unroll
  for (int i = 0; i < 8; ++i) tot += red[i];
  float rstd = rsqrtf(tot * (1.0f / (float)H_DIM) + 1e-6f);
  for (int i = threadIdx.x; i < H_DIM; i += 256) {
    float v = x[i] * rstd * w[i];
    out_bf[(size_t)row * H_DIM + i] = (bf16_t)v;
    if (F32OUT) out_f[(size_t)row * H_DIM + i] = v;
  }
}

// bf16 matrix transpose (rows x cols -> cols x rows), 32x32 LDS tiles
__global__ void __launch_bounds__(256)
transpose_bf16(const bf16_t* __restrict__ in, bf16_t* __restrict__ out,
               int rows, int cols) {
  __shared__ bf16_t tile[32][33];
  int tx = threadIdx.x & 31;
  int ty = threadIdx.x >> 5;      // 0..7
  int r0 = blockIdx.y * 32;
  int c0 = blockIdx.x * 32;
#pragma unroll
  for (int i = 0; i < 4; ++i) {
    int r = ty * 4 + i;
    tile[r][tx] = in[(size_t)(r0 + r) * cols + c0 + tx];
  }
  __syncthreads();
#pragma unroll
  for (int i = 0; i < 4; ++i) {
    int c = ty * 4 + i;
    out[(size_t)(c0 + c) * rows + r0 + tx] = tile[tx][c];
  }
}

__global__ void convert_f32_bf16(const float* __restrict__ src,
                                 bf16_t* __restrict__ dst, size_t n) {
  size_t i = (size_t)blockIdx.x * blockDim.x + threadIdx.x;
  size_t stride = (size_t)gridDim.x * blockDim.x;
  for (; i < n; i += stride) dst[i] = (bf16_t)src[i];
}

__global__ void copy_f32(const float* __restrict__ src, float* __restrict__ dst, size_t n) {
  size_t i = (size_t)blockIdx.x * blockDim.x + threadIdx.x;
  size_t stride = (size_t)gridDim.x * blockDim.x;
  for (; i < n; i += stride) dst[i] = src[i];
}

__global__ void silu_mul(const bf16_t* __restrict__ g, const bf16_t* __restrict__ u,
                         bf16_t* __restrict__ out, size_t n) {
  size_t i = (size_t)blockIdx.x * blockDim.x + threadIdx.x;
  size_t stride = (size_t)gridDim.x * blockDim.x;
  for (; i < n; i += stride) {
    float x = (float)g[i];
    float s = x / (1.f + __expf(-x));
    out[i] = (bf16_t)(s * (float)u[i]);
  }
}

// ---------------------------------------------------------------------------
extern "C" void kernel_launch(void* const* d_in, const int* in_sizes, int n_in,
                              void* d_out, int out_size, void* d_ws, size_t ws_size,
                              hipStream_t stream) {
  (void)in_sizes; (void)n_in; (void)out_size; (void)ws_size;

  const float* hidden      = (const float*)d_in[0];
  const float* q_w         = (const float*)d_in[1];
  const float* k_w         = (const float*)d_in[2];
  const float* v_w         = (const float*)d_in[3];
  const float* o_w         = (const float*)d_in[4];
  const float* attn_norm_w = (const float*)d_in[5];
  const float* mlp_norm_w  = (const float*)d_in[6];
  const float* gate_w      = (const float*)d_in[7];
  const float* up_w        = (const float*)d_in[8];
  const float* down_w      = (const float*)d_in[9];
  float* ct = (float*)d_out;                    // [S, H] fp32 running state

  char* base = (char*)d_ws;
  size_t off = 0;
  auto alloc = [&](size_t bytes) -> void* {
    void* p = base + off;
    off = (off + bytes + 255) & ~(size_t)255;
    return p;
  };

  const size_t KV = (size_t)NKV_ * HD_;         // 512
  bf16_t* qw  = (bf16_t*)alloc((size_t)H_DIM * H_DIM * 2);
  bf16_t* kw  = (bf16_t*)alloc(KV * H_DIM * 2);
  bf16_t* vw  = (bf16_t*)alloc(KV * H_DIM * 2);
  bf16_t* ow  = (bf16_t*)alloc((size_t)H_DIM * H_DIM * 2);
  bf16_t* gw  = (bf16_t*)alloc((size_t)I_DIM * H_DIM * 2);
  bf16_t* uw  = (bf16_t*)alloc((size_t)I_DIM * H_DIM * 2);
  bf16_t* dw  = (bf16_t*)alloc((size_t)H_DIM * I_DIM * 2);
  bf16_t* resid = (bf16_t*)alloc((size_t)TS_LEN * H_DIM * 2);
  bf16_t* xb  = (bf16_t*)alloc((size_t)S_LEN * H_DIM * 2);
  bf16_t* yb  = (bf16_t*)alloc((size_t)S_LEN * H_DIM * 2);
  float*  yf  = (float*) alloc((size_t)S_LEN * H_DIM * 4);
  bf16_t* qb  = (bf16_t*)alloc((size_t)S_LEN * H_DIM * 2);
  bf16_t* kb  = (bf16_t*)alloc((size_t)TS_LEN * KV * 2);
  bf16_t* vb  = (bf16_t*)alloc((size_t)TS_LEN * KV * 2);
  bf16_t* vTb = (bf16_t*)alloc(KV * (size_t)TS_LEN * 2);
  bf16_t* ob  = (bf16_t*)alloc((size_t)S_LEN * H_DIM * 2);
  bf16_t* gb  = (bf16_t*)alloc((size_t)S_LEN * I_DIM * 2);
  bf16_t* ub  = (bf16_t*)alloc((size_t)S_LEN * I_DIM * 2);
  bf16_t* hb  = (bf16_t*)alloc((size_t)S_LEN * I_DIM * 2);

  dim3 cvb(256);
  dim3 cvg(1024);
  convert_f32_bf16<<<cvg, cvb, 0, stream>>>(q_w,   qw, (size_t)H_DIM * H_DIM);
  convert_f32_bf16<<<cvg, cvb, 0, stream>>>(k_w,   kw, KV * H_DIM);
  convert_f32_bf16<<<cvg, cvb, 0, stream>>>(v_w,   vw, KV * H_DIM);
  convert_f32_bf16<<<cvg, cvb, 0, stream>>>(o_w,   ow, (size_t)H_DIM * H_DIM);
  convert_f32_bf16<<<cvg, cvb, 0, stream>>>(gate_w, gw, (size_t)I_DIM * H_DIM);
  convert_f32_bf16<<<cvg, cvb, 0, stream>>>(up_w,   uw, (size_t)I_DIM * H_DIM);
  convert_f32_bf16<<<cvg, cvb, 0, stream>>>(down_w, dw, (size_t)H_DIM * I_DIM);
  convert_f32_bf16<<<cvg, cvb, 0, stream>>>(hidden, resid, (size_t)T_LEN * H_DIM);
  // ct = hidden[:, -S:, :]
  copy_f32<<<cvg, cvb, 0, stream>>>(hidden + (size_t)S_LEN * H_DIM, ct, (size_t)S_LEN * H_DIM);

  dim3 blk(256);
  for (int depth = 0; depth < 2; ++depth) {
    convert_f32_bf16<<<cvg, cvb, 0, stream>>>(ct, resid + (size_t)T_LEN * H_DIM,
                                              (size_t)S_LEN * H_DIM);
    rmsnorm_kernel<false><<<dim3(S_LEN), blk, 0, stream>>>(ct, attn_norm_w, xb, nullptr);

    gemm_nt<false, true><<<dim3(H_DIM / 64, S_LEN / 128), blk, 0, stream>>>(
        xb, qw, nullptr, qb, S_LEN, H_DIM, H_DIM);
    gemm_nt<false, true><<<dim3((int)KV / 64, TS_LEN / 128), blk, 0, stream>>>(
        resid, kw, nullptr, kb, TS_LEN, (int)KV, H_DIM);
    gemm_nt<false, true><<<dim3((int)KV / 64, TS_LEN / 128), blk, 0, stream>>>(
        resid, vw, nullptr, vb, TS_LEN, (int)KV, H_DIM);
    transpose_bf16<<<dim3((int)KV / 32, TS_LEN / 32), blk, 0, stream>>>(
        vb, vTb, TS_LEN, (int)KV);

    attn_kernel<<<dim3(S_LEN / 16, NH_), dim3(32), 0, stream>>>(qb, kb, vTb, ob);

    gemm_nt<true, false><<<dim3(H_DIM / 64, S_LEN / 128), blk, 0, stream>>>(
        ob, ow, ct, ct, S_LEN, H_DIM, H_DIM);

    rmsnorm_kernel<true><<<dim3(S_LEN), blk, 0, stream>>>(ct, mlp_norm_w, yb, yf);

    gemm_nt<false, true><<<dim3(I_DIM / 64, S_LEN / 128), blk, 0, stream>>>(
        yb, gw, nullptr, gb, S_LEN, I_DIM, H_DIM);
    gemm_nt<false, true><<<dim3(I_DIM / 64, S_LEN / 128), blk, 0, stream>>>(
        yb, uw, nullptr, ub, S_LEN, I_DIM, H_DIM);
    silu_mul<<<cvg, cvb, 0, stream>>>(gb, ub, hb, (size_t)S_LEN * I_DIM);
    gemm_nt<true, false><<<dim3(H_DIM / 64, S_LEN / 128), blk, 0, stream>>>(
        hb, dw, yf, ct, S_LEN, H_DIM, I_DIM);
  }
}
